// RNNEncoder3D_35253091565903
// MI455X (gfx1250) — matrix operand order repaired and checked by reading, not compiled, not run on hardware
//
#include <hip/hip_runtime.h>
#include <hip/hip_bf16.h>

typedef _Float16 v16h __attribute__((ext_vector_type(16)));
typedef _Float16 v8h  __attribute__((ext_vector_type(8)));
typedef float    v8f  __attribute__((ext_vector_type(8)));

constexpr int B_ = 32, T_ = 1024, C_ = 512, H_ = 512, G_ = 2048; // G = 4H
constexpr int NWG = 32;        // persistent workgroups in recurrent phase
constexpr int J_  = H_ / NWG;  // 16 hidden units per WG (one 16-wide N tile per gate)

static __device__ __forceinline__ v16h cat16(v8h lo, v8h hi) {
  return __builtin_shufflevector(lo, hi, 0,1,2,3,4,5,6,7,8,9,10,11,12,13,14,15);
}

static __device__ __forceinline__ float sigmoid_fast(float x) {
  return 1.f / (1.f + __expf(-x));           // v_exp_f32
}
static __device__ __forceinline__ float tanh_fast(float x) {
  return 1.f - 2.f / (__expf(2.f * x) + 1.f); // exact limits: +inf->1, 0->-1
}

// ---- prep: X [B,T,C] f32 -> Xf16 [T,B,C] f16 (rows match GEMM row r = t*B+b) ----
__global__ void cvt_x_kernel(const float* __restrict__ x, _Float16* __restrict__ xf) {
  int o = blockIdx.x * blockDim.x + threadIdx.x;   // index into [T][B][C]
  int t = o / (B_ * C_);
  int r = o % (B_ * C_);
  int b = r / C_;
  int c = r % C_;
  xf[o] = (_Float16)x[(size_t)b * T_ * C_ + (size_t)t * C_ + c];
}

// ---- prep: W [G,512] f32 -> Wt [512,G] f16 (K-major for B fragments) ----
__global__ void cvt_wT_kernel(const float* __restrict__ w, _Float16* __restrict__ wt) {
  int o = blockIdx.x * blockDim.x + threadIdx.x;   // index into [K=512][G]
  int k = o / G_;
  int n = o % G_;
  wt[o] = (_Float16)w[(size_t)n * 512 + k];
}

// ---- phase 1: Xp[r][n] = sum_k Xf16[r][k]*WihT[k][n] + b_ih[n] + b_hh[n] ----
// wave computes 2 M-tiles x 4 N-tiles; block = 8 waves stacked in M (256 rows x 64 cols)
__global__ void __launch_bounds__(256) xproj_gemm_kernel(
    const _Float16* __restrict__ A,   // [M=T*B][512]
    const _Float16* __restrict__ Bt,  // [512][G]
    const float* __restrict__ b_ih, const float* __restrict__ b_hh,
    float* __restrict__ Xp)           // [M][G]
{
  const int lane  = threadIdx.x & 31;
  const int wave  = threadIdx.x >> 5;
  const int mbase = blockIdx.x * 256 + wave * 32;
  const int nbase = blockIdx.y * 64;
  const int mrow  = lane & 15;
  const int koff  = (lane >> 4) * 8;

  v8f zero = {0.f,0.f,0.f,0.f,0.f,0.f,0.f,0.f};
  v8f acc[2][4];
  for (int i = 0; i < 2; ++i)
    for (int j = 0; j < 4; ++j) acc[i][j] = zero;

  for (int k0 = 0; k0 < C_; k0 += 32) {
    v16h a[2];
#pragma unroll
    for (int mt = 0; mt < 2; ++mt) {
      const _Float16* p = A + (size_t)(mbase + mt * 16 + mrow) * C_ + k0 + koff;
      a[mt] = cat16(*(const v8h*)p, *(const v8h*)(p + 16));
    }
    const _Float16* q = Bt + (size_t)(k0 + koff) * G_ + nbase + mrow;
#pragma unroll
    for (int nt = 0; nt < 4; ++nt) {
      v16h b;
#pragma unroll
      for (int s = 0; s < 8; ++s) {
        b[s]     = q[(size_t)s * G_ + nt * 16];
        b[s + 8] = q[(size_t)(s + 16) * G_ + nt * 16];
      }
#pragma unroll
      for (int mt = 0; mt < 2; ++mt)
        acc[mt][nt] = __builtin_amdgcn_wmma_f32_16x16x32_f16(
            false, a[mt], false, b, (short)0, acc[mt][nt], false, false);
    }
  }

  const int drow = (lane >> 4) * 8;
  const int dcol = lane & 15;
#pragma unroll
  for (int nt = 0; nt < 4; ++nt) {
    int n = nbase + nt * 16 + dcol;
    float bias = b_ih[n] + b_hh[n];
#pragma unroll
    for (int mt = 0; mt < 2; ++mt) {
#pragma unroll
      for (int r = 0; r < 8; ++r) {
        int R = mbase + mt * 16 + drow + r;
        // written once, consumed once much later: keep it out of the caches
        __builtin_nontemporal_store(acc[mt][nt][r] + bias, &Xp[(size_t)R * G_ + n]);
      }
    }
  }
}

// ---- phase 2: persistent recurrent kernel, NWG workgroups, grid barrier/step ----
__global__ void __launch_bounds__(256) lstm_rec_kernel(
    const float* __restrict__ Xp,        // [T*B][G]
    const _Float16* __restrict__ WhhT,   // [512][G]
    const int* __restrict__ lens,        // [B]
    _Float16* __restrict__ hbuf,         // [2][B][H] f16, zero-initialized
    unsigned* __restrict__ bar,          // [0]=count [1]=sense, zero-initialized
    float* __restrict__ out)             // outs[T][B][H], hT[B][H], cT[B][H]
{
  __shared__ _Float16 lds_w[4 * 512 * J_];   // Whh^T slice: [gc][k][jj]  (64 KB)
  __shared__ float    gates[B_ * 4 * J_];    // [b][gc*J_+jj]             (8 KB)
  __shared__ float    c_st[B_ * J_];
  __shared__ float    h_st[B_ * J_];

  const int tid  = threadIdx.x;
  const int lane = tid & 31;
  const int wave = tid >> 5;
  const int wg   = blockIdx.x;
  const int j0   = wg * J_;

  // stage this WG's 64 gate columns of Whh^T into LDS (once)
  for (int idx = tid; idx < 4 * 512 * J_; idx += 256) {
    int gc  = idx / (512 * J_);
    int rem = idx % (512 * J_);
    int k   = rem / J_;
    int jj  = rem % J_;
    lds_w[idx] = WhhT[(size_t)k * G_ + gc * H_ + j0 + jj];
  }
  for (int idx = tid; idx < B_ * J_; idx += 256) { c_st[idx] = 0.f; h_st[idx] = 0.f; }
  __syncthreads();

  // wave -> output tile: mt in {0,1} (batch rows), gc in {0..3} (i,f,g,o)
  const int mt   = wave >> 2;
  const int gc   = wave & 3;
  const int mrow = lane & 15;
  const int koff = (lane >> 4) * 8;
  const int drow = (lane >> 4) * 8;
  const int dcol = lane & 15;

  // per-thread elementwise assignment (fixed across time)
  int pb[2], pj[2], plen[2];
#pragma unroll
  for (int q = 0; q < 2; ++q) {
    int p = tid + q * 256;
    pb[q] = p >> 4;           // J_ = 16
    pj[q] = p & (J_ - 1);
    plen[q] = lens[pb[q]];
  }

  // x_proj gate preactivations for the current step, prefetched into registers
  float xv[2][4];
  {
#pragma unroll
    for (int q = 0; q < 2; ++q) {
      const float* xr = Xp + (size_t)pb[q] * G_ + j0 + pj[q];  // t = 0
#pragma unroll
      for (int g = 0; g < 4; ++g) xv[q][g] = __builtin_nontemporal_load(xr + g * 512);
    }
  }

  v8f zero = {0.f,0.f,0.f,0.f,0.f,0.f,0.f,0.f};

  for (int t = 0; t < T_; ++t) {
    const _Float16* hb_in  = hbuf + (size_t)(t & 1) * B_ * H_;
    _Float16*       hb_out = hbuf + (size_t)((t + 1) & 1) * B_ * H_;

    // gates_tile = h @ Whh^T (slice), K = 512 in 16 chunks
    v8f acc = zero;
    const _Float16* arow = hb_in + (size_t)(mt * 16 + mrow) * H_ + koff;
#pragma unroll
    for (int k0 = 0; k0 < H_; k0 += 32) {
      v16h a = cat16(*(const v8h*)(arow + k0), *(const v8h*)(arow + k0 + 16));
      const _Float16* brow = &lds_w[((size_t)gc * 512 + k0 + koff) * J_ + dcol];
      v16h b;
#pragma unroll
      for (int s = 0; s < 8; ++s) {
        b[s]     = brow[s * J_];
        b[s + 8] = brow[(s + 16) * J_];
      }
      acc = __builtin_amdgcn_wmma_f32_16x16x32_f16(
          false, a, false, b, (short)0, acc, false, false);
    }
#pragma unroll
    for (int r = 0; r < 8; ++r) {
      int row = mt * 16 + drow + r;
      gates[row * (4 * J_) + gc * J_ + dcol] = acc[r];
    }
    __syncthreads();

    // elementwise LSTM cell for this WG's hidden slice
#pragma unroll
    for (int q = 0; q < 2; ++q) {
      int b  = pb[q];
      int jj = pj[q];
      float iv = sigmoid_fast(gates[b * 4 * J_ + 0 * J_ + jj] + xv[q][0]);
      float fv = sigmoid_fast(gates[b * 4 * J_ + 1 * J_ + jj] + xv[q][1]);
      float gv = tanh_fast  (gates[b * 4 * J_ + 2 * J_ + jj] + xv[q][2]);
      float ov = sigmoid_fast(gates[b * 4 * J_ + 3 * J_ + jj] + xv[q][3]);
      float cold = c_st[b * J_ + jj];
      float hold = h_st[b * J_ + jj];
      float cnew = fv * cold + iv * gv;
      float hnew = ov * tanh_fast(cnew);
      bool valid = t < plen[q];
      float ck = valid ? cnew : cold;
      float hk = valid ? hnew : hold;
      c_st[b * J_ + jj] = ck;
      h_st[b * J_ + jj] = hk;
      hb_out[(size_t)b * H_ + j0 + jj] = (_Float16)hk;
      __builtin_nontemporal_store(valid ? hnew : 0.f,
                                  &out[((size_t)t * B_ + b) * H_ + j0 + jj]);
    }

    // device-wide barrier, with next step's x_proj prefetch hidden under the wait
    __builtin_amdgcn_fence(__ATOMIC_RELEASE, "agent");  // drain h stores to L2
    __syncthreads();
    if (tid == 0) {
      unsigned arrived = __hip_atomic_fetch_add(&bar[0], 1u, __ATOMIC_RELAXED,
                                                __HIP_MEMORY_SCOPE_AGENT) + 1u;
      if (arrived == (unsigned)NWG) {
        __hip_atomic_store(&bar[0], 0u, __ATOMIC_RELAXED, __HIP_MEMORY_SCOPE_AGENT);
        __hip_atomic_store(&bar[1], (unsigned)(t + 1), __ATOMIC_RELEASE,
                           __HIP_MEMORY_SCOPE_AGENT);
      }
    }
    if (t + 1 < T_) {   // issue loads now; they complete while we wait
#pragma unroll
      for (int q = 0; q < 2; ++q) {
        const float* xr = Xp + ((size_t)(t + 1) * B_ + pb[q]) * G_ + j0 + pj[q];
#pragma unroll
        for (int g = 0; g < 4; ++g) xv[q][g] = __builtin_nontemporal_load(xr + g * 512);
      }
    }
    if (tid == 0) {
      while (__hip_atomic_load(&bar[1], __ATOMIC_ACQUIRE,
                               __HIP_MEMORY_SCOPE_AGENT) < (unsigned)(t + 1)) {
        __builtin_amdgcn_s_sleep(1);
      }
    }
    __syncthreads();
    __builtin_amdgcn_fence(__ATOMIC_ACQUIRE, "agent");  // invalidate WGP$ for fresh h
  }

  // final hidden / cell state
  float* hT = out + (size_t)T_ * B_ * H_;
  float* cT = hT + (size_t)B_ * H_;
#pragma unroll
  for (int q = 0; q < 2; ++q) {
    int b  = pb[q];
    int jj = pj[q];
    hT[(size_t)b * H_ + j0 + jj] = h_st[b * J_ + jj];
    cT[(size_t)b * H_ + j0 + jj] = c_st[b * J_ + jj];
  }
}

extern "C" void kernel_launch(void* const* d_in, const int* in_sizes, int n_in,
                              void* d_out, int out_size, void* d_ws, size_t ws_size,
                              hipStream_t stream) {
  const float* x    = (const float*)d_in[0];
  const int*   lens = (const int*)d_in[1];
  const float* W_ih = (const float*)d_in[2];
  const float* W_hh = (const float*)d_in[3];
  const float* b_ih = (const float*)d_in[4];
  const float* b_hh = (const float*)d_in[5];
  float* out = (float*)d_out;

  // workspace layout
  char* ws = (char*)d_ws;
  const size_t o_bar  = 0;                                   // 256 B (8 used)
  const size_t o_hbuf = 256;                                 // 2*B*H f16 = 64 KB
  const size_t o_xf   = o_hbuf + (size_t)2 * B_ * H_ * 2;    // T*B*C f16 = 32 MB
  const size_t o_wih  = o_xf + (size_t)T_ * B_ * C_ * 2;     // 512*G f16 = 2 MB
  const size_t o_whh  = o_wih + (size_t)512 * G_ * 2;        // 512*G f16 = 2 MB
  const size_t o_xp   = o_whh + (size_t)512 * G_ * 2;        // T*B*G f32 = 256 MB
  const size_t needed = o_xp + (size_t)T_ * B_ * G_ * 4;
  if (ws_size < needed) return;

  unsigned* bar  = (unsigned*)(ws + o_bar);
  _Float16* hbuf = (_Float16*)(ws + o_hbuf);
  _Float16* xf16 = (_Float16*)(ws + o_xf);
  _Float16* wihT = (_Float16*)(ws + o_wih);
  _Float16* whhT = (_Float16*)(ws + o_whh);
  float*    Xp   = (float*)(ws + o_xp);

  hipMemsetAsync(ws, 0, o_xf, stream);  // zero barrier + h double-buffer

  cvt_x_kernel<<<(T_ * B_ * C_) / 256, 256, 0, stream>>>(x, xf16);
  cvt_wT_kernel<<<(512 * G_) / 256, 256, 0, stream>>>(W_ih, wihT);
  cvt_wT_kernel<<<(512 * G_) / 256, 256, 0, stream>>>(W_hh, whhT);

  dim3 g1((T_ * B_) / 256, G_ / 64);
  xproj_gemm_kernel<<<g1, 256, 0, stream>>>(xf16, wihT, b_ih, b_hh, Xp);

  lstm_rec_kernel<<<NWG, 256, 0, stream>>>(Xp, whhT, lens, hbuf, bar, out);
}
